// HeteroGAT_82308753261157
// MI455X (gfx1250) — matrix operand order repaired and checked by reading, not compiled, not run on hardware
//
#include <hip/hip_runtime.h>

typedef __attribute__((ext_vector_type(16))) _Float16 v16h;
typedef __attribute__((ext_vector_type(4)))  _Float16 v4h;
typedef __attribute__((ext_vector_type(8)))  float    v8f;

#define NEG_SLOPE 0.2f

__device__ __forceinline__ unsigned f2ord(float f) {
    unsigned b = __float_as_uint(f);
    return (b & 0x80000000u) ? ~b : (b | 0x80000000u);
}
__device__ __forceinline__ float ord2f(unsigned u) {
    return (u & 0x80000000u) ? __uint_as_float(u & 0x7fffffffu)
                             : __uint_as_float(~u);
}

// ---------------------------------------------------------------------------
// GEMM: Z[n, NOUT] = X[n, 128] @ W[128, NOUT]  (fp32 in/out, f16 WMMA compute)
// Block = 256 threads = 8 waves; each wave owns 16 rows x NOUT cols.
// ---------------------------------------------------------------------------
template <int NOUT>
__global__ __launch_bounds__(256)
void gemm_wmma_k(const float* __restrict__ X, const float* __restrict__ W,
                 float* __restrict__ Z, int nrows) {
    constexpr int DIN = 128;
    constexpr int KT  = 32;
    constexpr int NT  = NOUT / 16;
    __shared__ __align__(16) _Float16 lx[128][KT];   // x tile, row-major
    __shared__ __align__(16) _Float16 lw[NOUT][KT];  // W tile, transposed [n][k]

    const int t    = threadIdx.x;
    const int wave = t >> 5;
    const int lane = t & 31;
    const int nm   = lane & 15;   // A: row M; B: col N; C/D: col N
    const int hfw  = lane >> 4;   // half-wave select
    const int rowBlock = blockIdx.x * 128;
    const int waveRow0 = rowBlock + wave * 16;

    v8f zero = {0.f,0.f,0.f,0.f,0.f,0.f,0.f,0.f};
    v8f acc[NT];
#pragma unroll
    for (int i = 0; i < NT; ++i) acc[i] = zero;

    for (int k0 = 0; k0 < DIN; k0 += KT) {
        // stage X tile (128 rows x 32 k) as f16, coalesced float4 reads
        for (int e4 = t; e4 < 128 * KT / 4; e4 += 256) {
            int r  = e4 >> 3;
            int kc = (e4 & 7) * 4;
            float4 v = {0.f, 0.f, 0.f, 0.f};
            int gr = rowBlock + r;
            if (gr < nrows) v = *(const float4*)(X + (size_t)gr * DIN + k0 + kc);
            v4h hv = { (_Float16)v.x, (_Float16)v.y, (_Float16)v.z, (_Float16)v.w };
            *(v4h*)&lx[r][kc] = hv;
        }
        // stage W tile (32 k x NOUT n), stored transposed lw[n][k]
        for (int e4 = t; e4 < NOUT * KT / 4; e4 += 256) {
            int k  = e4 / (NOUT / 4);
            int nc = (e4 % (NOUT / 4)) * 4;
            float4 v = *(const float4*)(W + (size_t)(k0 + k) * NOUT + nc);
            lw[nc + 0][k] = (_Float16)v.x;
            lw[nc + 1][k] = (_Float16)v.y;
            lw[nc + 2][k] = (_Float16)v.z;
            lw[nc + 3][k] = (_Float16)v.w;
        }
        __syncthreads();
        if (waveRow0 < nrows) {   // wave-uniform: EXEC stays all-ones for WMMA
            // A operand: 16x32, lanes 0-15 K=0..15, lanes 16-31 K=16..31 (matches B)
            union { v16h v; unsigned u[8]; } A;
            const unsigned* xr = (const unsigned*)&lx[nm][0];
#pragma unroll
            for (int i = 0; i < 8; ++i) {
                A.u[i]     = xr[hfw * 8 + i];
            }
#pragma unroll
            for (int nt = 0; nt < NT; ++nt) {
                // B operand: 32x16, lanes 0-15 K=0..15, lanes 16-31 K=16..31
                union { v16h v; unsigned u[8]; } B;
                const unsigned* wr = (const unsigned*)&lw[nt * 16 + nm][0];
#pragma unroll
                for (int i = 0; i < 8; ++i) B.u[i] = wr[hfw * 8 + i];
                acc[nt] = __builtin_amdgcn_wmma_f32_16x16x32_f16(
                    false, A.v, false, B.v, (short)0, acc[nt], false, false);
            }
        }
        __syncthreads();
    }
    if (waveRow0 < nrows) {
#pragma unroll
        for (int nt = 0; nt < NT; ++nt)
#pragma unroll
            for (int v = 0; v < 8; ++v) {
                int row = waveRow0 + hfw * 8 + v;   // C/D layout: M = v + 8*half
                Z[(size_t)row * NOUT + nt * 16 + nm] = acc[nt][v];
            }
    }
}

// ---------------------------------------------------------------------------
// el/er = einsum('nhf,hf->nh'): one thread per (node, head)
// ---------------------------------------------------------------------------
template <int H, int F>
__global__ void attn_proj_k(const float* __restrict__ z, const float* __restrict__ al,
                            const float* __restrict__ ar, float* __restrict__ el,
                            float* __restrict__ er, int n) {
    int tid = blockIdx.x * blockDim.x + threadIdx.x;
    if (tid >= n * H) return;
    int node = tid / H, h = tid % H;
    const float* zr  = z + (size_t)node * (H * F) + h * F;
    const float* alh = al + h * F;
    const float* arh = ar + h * F;
    float sl = 0.f, sr = 0.f;
#pragma unroll
    for (int f = 0; f < F; ++f) { sl += zr[f] * alh[f]; sr += zr[f] * arh[f]; }
    el[tid] = sl;
    er[tid] = sr;
}

// ---------------------------------------------------------------------------
// Edge pass A: logits + LeakyReLU, stash to pbuf, segment-max via ordered u32
// ---------------------------------------------------------------------------
template <int H>
__global__ void edge_logits_max_k(const int* __restrict__ src, const int* __restrict__ dst,
                                  const float* __restrict__ el, const float* __restrict__ er,
                                  float* __restrict__ pbuf, unsigned* __restrict__ mbuf, int E) {
    int tid = blockIdx.x * blockDim.x + threadIdx.x;
    if (tid >= E * H) return;
    int e = tid / H, h = tid - e * H;
    int s = src[e], d = dst[e];
    float v = el[s * H + h] + er[d * H + h];
    v = (v > 0.f) ? v : NEG_SLOPE * v;
    pbuf[tid] = v;
    atomicMax(&mbuf[d * H + h], f2ord(v));
}

// Edge pass B: p = exp(e - m[dst]), segment-sum denom
template <int H>
__global__ void edge_expsum_k(const int* __restrict__ dst, float* __restrict__ pbuf,
                              const unsigned* __restrict__ mbuf, float* __restrict__ denom, int E) {
    int tid = blockIdx.x * blockDim.x + threadIdx.x;
    if (tid >= E * H) return;
    int e = tid / H, h = tid - e * H;
    int d = dst[e];
    float m = ord2f(mbuf[d * H + h]);
    float p = __expf(pbuf[tid] - m);
    pbuf[tid] = p;
    atomicAdd(&denom[d * H + h], p);
}

// Edge pass C: agg[dst] += alpha * z[src]; thread = (edge, 4-channel chunk)
template <int H, int F>
__global__ void edge_aggregate_k(const int* __restrict__ src, const int* __restrict__ dst,
                                 const float* __restrict__ pbuf, const float* __restrict__ denom,
                                 const float* __restrict__ z, float* __restrict__ agg, int E) {
    constexpr int CH = H * F;
    constexpr int CHUNKS = CH / 4;
    int tid = blockIdx.x * blockDim.x + threadIdx.x;
    if (tid >= E * CHUNKS) return;
    int e = tid / CHUNKS;
    int j = tid - e * CHUNKS;
    int c = j * 4;
    int h = c / F;
    int s = src[e], d = dst[e];
    float alpha = pbuf[e * H + h] / denom[d * H + h];
    float4 zv = *(const float4*)(z + (size_t)s * CH + c);
    float* o = agg + (size_t)d * CH + c;
    atomicAdd(o + 0, alpha * zv.x);
    atomicAdd(o + 1, alpha * zv.y);
    atomicAdd(o + 2, alpha * zv.z);
    atomicAdd(o + 3, alpha * zv.w);
}

__global__ void fill_u32_k(unsigned* __restrict__ p, unsigned v, int n) {
    int i = blockIdx.x * blockDim.x + threadIdx.x;
    if (i < n) p[i] = v;
}

// h1 = elu((sum_r agg_r + sum_r b1_r) / 3), in place
__global__ void finalize1_k(float* __restrict__ agg, const float* __restrict__ b1, int n) {
    int tid = blockIdx.x * blockDim.x + threadIdx.x;
    if (tid >= n * 128) return;
    int c = tid & 127;
    float bs = b1[c] + b1[128 + c] + b1[256 + c];
    float v = (agg[tid] + bs) * (1.f / 3.f);
    agg[tid] = (v > 0.f) ? v : (__expf(v) - 1.f);
}

__global__ void finalize2_k(const float* __restrict__ agg, const float* __restrict__ b2,
                            float* __restrict__ out, int n) {
    int tid = blockIdx.x * blockDim.x + threadIdx.x;
    if (tid >= n * 16) return;
    int c = tid & 15;
    float bs = b2[c] + b2[16 + c] + b2[32 + c];
    out[tid] = (agg[tid] + bs) * (1.f / 3.f);
}

// ---------------------------------------------------------------------------
extern "C" void kernel_launch(void* const* d_in, const int* in_sizes, int n_in,
                              void* d_out, int out_size, void* d_ws, size_t ws_size,
                              hipStream_t stream) {
    (void)in_sizes; (void)n_in; (void)out_size; (void)ws_size;
    const float* x   = (const float*)d_in[0];
    const int*   src = (const int*)d_in[1];
    const int*   dst = (const int*)d_in[2];
    const float* W1  = (const float*)d_in[3];
    const float* al1 = (const float*)d_in[4];
    const float* ar1 = (const float*)d_in[5];
    const float* b1  = (const float*)d_in[6];
    const float* W2  = (const float*)d_in[7];
    const float* al2 = (const float*)d_in[8];
    const float* ar2 = (const float*)d_in[9];
    const float* b2  = (const float*)d_in[10];
    float* out = (float*)d_out;

    constexpr int N = 100000, H = 4, F = 32, C = 16, R = 3, E = 1600000;
    constexpr int D = H * F;  // 128

    char* p = (char*)d_ws;
    float*    z    = (float*)p;    p += (size_t)N * D * sizeof(float);
    float*    aggs = (float*)p;    p += (size_t)N * D * sizeof(float);
    float*    el   = (float*)p;    p += (size_t)N * H * sizeof(float);
    float*    er   = (float*)p;    p += (size_t)N * H * sizeof(float);
    unsigned* mb   = (unsigned*)p; p += (size_t)N * H * sizeof(unsigned);
    float*    den  = (float*)p;    p += (size_t)N * H * sizeof(float);
    float*    pb   = (float*)p;    p += (size_t)E * H * sizeof(float);
    float*    agg2 = (float*)p;    p += (size_t)N * C * sizeof(float);

    const unsigned NEGINF_ORD = 0x007FFFFFu;  // f2ord(-inf)
    auto cdiv = [](long long a, long long b) { return (int)((a + b - 1) / b); };

    // ----- layer 1: per-relation GAT(128 -> 4x32), mean over relations -----
    hipMemsetAsync(aggs, 0, (size_t)N * D * sizeof(float), stream);
    for (int r = 0; r < R; ++r) {
        gemm_wmma_k<128><<<cdiv(N, 128), 256, 0, stream>>>(
            x, W1 + (size_t)r * 128 * 128, z, N);
        attn_proj_k<4, 32><<<cdiv((long long)N * H, 256), 256, 0, stream>>>(
            z, al1 + r * H * F, ar1 + r * H * F, el, er, N);
        fill_u32_k<<<cdiv((long long)N * H, 256), 256, 0, stream>>>(mb, NEGINF_ORD, N * H);
        hipMemsetAsync(den, 0, (size_t)N * H * sizeof(float), stream);
        edge_logits_max_k<4><<<cdiv((long long)E * H, 256), 256, 0, stream>>>(
            src + (size_t)r * E, dst + (size_t)r * E, el, er, pb, mb, E);
        edge_expsum_k<4><<<cdiv((long long)E * H, 256), 256, 0, stream>>>(
            dst + (size_t)r * E, pb, mb, den, E);
        edge_aggregate_k<4, 32><<<cdiv((long long)E * H * F / 4, 256), 256, 0, stream>>>(
            src + (size_t)r * E, dst + (size_t)r * E, pb, den, z, aggs, E);
    }
    finalize1_k<<<cdiv((long long)N * D, 256), 256, 0, stream>>>(aggs, b1, N);
    const float* h1 = aggs;  // [N, 128], ELU applied in place

    // ----- layer 2: per-relation GAT(128 -> 1x16), mean over relations -----
    hipMemsetAsync(agg2, 0, (size_t)N * C * sizeof(float), stream);
    for (int r = 0; r < R; ++r) {
        gemm_wmma_k<16><<<cdiv(N, 128), 256, 0, stream>>>(
            h1, W2 + (size_t)r * 128 * C, z, N);
        attn_proj_k<1, 16><<<cdiv(N, 256), 256, 0, stream>>>(
            z, al2 + r * C, ar2 + r * C, el, er, N);
        fill_u32_k<<<cdiv(N, 256), 256, 0, stream>>>(mb, NEGINF_ORD, N);
        hipMemsetAsync(den, 0, (size_t)N * sizeof(float), stream);
        edge_logits_max_k<1><<<cdiv(E, 256), 256, 0, stream>>>(
            src + (size_t)r * E, dst + (size_t)r * E, el, er, pb, mb, E);
        edge_expsum_k<1><<<cdiv(E, 256), 256, 0, stream>>>(
            dst + (size_t)r * E, pb, mb, den, E);
        edge_aggregate_k<1, 16><<<cdiv((long long)E * C / 4, 256), 256, 0, stream>>>(
            src + (size_t)r * E, dst + (size_t)r * E, pb, den, z, agg2, E);
    }
    finalize2_k<<<cdiv((long long)N * C, 256), 256, 0, stream>>>(agg2, b2, out, N);
}
